// CRF_32744830665080
// MI455X (gfx1250) — compile-verified, matrix operand order checked by reading, not run on hardware
//
#include <hip/hip_runtime.h>
#include <hip/hip_bf16.h>
#include <stdint.h>

// ---------------- problem constants ----------------
#define BDIM   64
#define LDIM   512
#define DDIM   1024
#define CTAGS  131          // NUM_TAGS + 3
#define FPAD   132          // padded row stride (tags)
#define START_IDX 129
#define STOP_IDX  130
#define NEGV  (-10000.0f)

#define NTILES 9            // ceil(131/16)
#define MROWS  (BDIM*LDIM)  // 32768

typedef __attribute__((ext_vector_type(16))) __bf16    v16bf;
typedef __attribute__((ext_vector_type(8)))  __bf16    v8bf;
typedef __attribute__((ext_vector_type(8)))  float     v8f;
typedef __attribute__((ext_vector_type(4)))  float     v4f;
typedef __attribute__((ext_vector_type(8)))  uint32_t  v8u;

// float -> bf16, round to nearest even (tiny W pre-pass only)
static __device__ inline uint16_t f2bf_rtne(float f) {
    uint32_t u = __builtin_bit_cast(uint32_t, f);
    uint32_t r = u + 0x7FFFu + ((u >> 16) & 1u);
    return (uint16_t)(r >> 16);
}

// pack two floats into one dword of two bf16 (round-half-up + v_perm_b32)
static __device__ inline uint32_t pkbf(float f0, float f1) {
    uint32_t u0 = __builtin_bit_cast(uint32_t, f0) + 0x8000u;
    uint32_t u1 = __builtin_bit_cast(uint32_t, f1) + 0x8000u;
    return __builtin_amdgcn_perm(u1, u0, 0x07060302u);
}

// =====================================================================
// Kernel 0: W fp32 [131,1024] -> bf16 [131,1024] (row-major, RTNE)
// =====================================================================
__global__ __launch_bounds__(256) void crf_convW(
    const float* __restrict__ W, uint16_t* __restrict__ wb)
{
    int i = blockIdx.x * blockDim.x + threadIdx.x;
    if (i < CTAGS * DDIM) wb[i] = f2bf_rtne(W[i]);
}

// =====================================================================
// Kernel 1: feats = x @ W^T + b via v_wmma_f32_16x16x32_bf16.
// One wave per 16-row strip, all 9 column tiles per wave. Per K-step:
// hoist all 18 B loads into 9 distinct register buffers (partial waits
// overlap loads with the WMMA chain), x loads are non-temporal.
// =====================================================================
__global__ __launch_bounds__(256) void crf_proj_wmma(
    const float*    __restrict__ x,      // [32768, 1024] f32
    const uint16_t* __restrict__ wb,     // [131, 1024] bf16
    const float*    __restrict__ bias,   // [131]
    float*          __restrict__ feats)  // [32768, FPAD]
{
    const int lane  = threadIdx.x & 31;
    const int wave  = threadIdx.x >> 5;
    const int mtile = blockIdx.x * 8 + wave;     // 0..2047
    const int row0  = mtile * 16;
    const int m     = lane & 15;                 // A row / B column in tile
    const int kp    = (lane >> 4) * 8;           // K sub-offset: 0 or 8

    const float* __restrict__ xrow = x + (size_t)(row0 + m) * DDIM + kp;

    const uint16_t* wrow[NTILES];
#pragma unroll
    for (int n = 0; n < NTILES; ++n) {
        int col = n * 16 + m;
        if (col > CTAGS - 1) col = CTAGS - 1;    // clamp; masked at store
        wrow[n] = wb + (size_t)col * DDIM + kp;
    }

    v8f acc[NTILES];
#pragma unroll
    for (int n = 0; n < NTILES; ++n)
        acc[n] = (v8f){0.f,0.f,0.f,0.f,0.f,0.f,0.f,0.f};

    for (int k0 = 0; k0 < DDIM; k0 += 32) {
        // ---- issue ALL B-fragment loads up front (distinct buffers)
        v8bf blo[NTILES], bhi[NTILES];
#pragma unroll
        for (int n = 0; n < NTILES; ++n) {
            blo[n] = *(const v8bf*)(wrow[n] + k0);
            bhi[n] = *(const v8bf*)(wrow[n] + k0 + 16);
        }
        // ---- A fragment: stream x (read-once) non-temporally
        v4f x0 = __builtin_nontemporal_load((const v4f*)(xrow + k0));
        v4f x1 = __builtin_nontemporal_load((const v4f*)(xrow + k0 + 4));
        v4f x2 = __builtin_nontemporal_load((const v4f*)(xrow + k0 + 16));
        v4f x3 = __builtin_nontemporal_load((const v4f*)(xrow + k0 + 20));
        v8u au;
        au[0] = pkbf(x0[0], x0[1]);
        au[1] = pkbf(x0[2], x0[3]);
        au[2] = pkbf(x1[0], x1[1]);
        au[3] = pkbf(x1[2], x1[3]);
        au[4] = pkbf(x2[0], x2[1]);
        au[5] = pkbf(x2[2], x2[3]);
        au[6] = pkbf(x3[0], x3[1]);
        au[7] = pkbf(x3[2], x3[3]);
        v16bf a = __builtin_bit_cast(v16bf, au);

#pragma unroll
        for (int n = 0; n < NTILES; ++n) {
            v16bf bm = __builtin_shufflevector(blo[n], bhi[n],
                0,1,2,3,4,5,6,7,8,9,10,11,12,13,14,15);
            acc[n] = __builtin_amdgcn_wmma_f32_16x16x32_bf16(
                false, a, false, bm, (short)0, acc[n], false, false);
        }
    }

    // ---- epilogue: C/D layout -> VGPR r: M = r + 8*(lane>=16), N = lane&15
    const int mbase = (lane >> 4) * 8;
    const int ncol  = lane & 15;
#pragma unroll
    for (int n = 0; n < NTILES; ++n) {
        int c = n * 16 + ncol;
        if (c < CTAGS) {
            float bb = bias[c];
#pragma unroll
            for (int r = 0; r < 8; ++r) {
                int row = row0 + mbase + r;
                feats[(size_t)row * FPAD + c] = acc[n][r] + bb;
            }
        }
    }
}

// =====================================================================
// Kernel 2: Viterbi scan + argmax + backtrack, fully LDS-resident.
//  - transitions padded to [131][132] f32 (69.2 KB)
//  - score vector padded to 132 (pad = -1e30 so it never wins argmax)
//  - backpointers as u8 in LDS: [512][132] = 67.5 KB (tags < 256)
// Total dynamic LDS ~137 KB (CDNA5 WGP: 320 KB). Inner max-plus runs as
// 33 x b128 LDS loads + vec4 compares; backtrack is all-LDS.
// =====================================================================
__global__ __launch_bounds__(160) void crf_viterbi(
    const float* __restrict__ feats,  // [32768, FPAD]
    const float* __restrict__ mask,   // [64, 512]
    const float* __restrict__ trans,  // [131, 131]
    float*       __restrict__ out)    // [64] scores + [64*512] paths
{
    extern __shared__ float smem[];
    float*   s_trans = smem;                        // 131*132
    float*   s_score = smem + CTAGS * FPAD;         // 132
    uint8_t* s_bp    = (uint8_t*)(s_score + FPAD);  // 512*132

    const int b   = blockIdx.x;
    const int tid = threadIdx.x;
    const int c   = tid;
    const int cc  = (c < CTAGS) ? c : (CTAGS - 1);

    // transitions -> LDS, padded column = 0
    for (int i = tid; i < CTAGS * FPAD; i += blockDim.x) {
        int r  = i / FPAD;
        int cx = i - r * FPAD;
        s_trans[i] = (cx < CTAGS) ? trans[r * CTAGS + cx] : 0.0f;
    }
    if (c < CTAGS) s_score[c] = (c == START_IDX) ? 0.0f : NEGV;
    if (c == CTAGS) s_score[CTAGS] = -1.0e30f;      // pad: never selected
    __syncthreads();

    const float* __restrict__ trow = s_trans + (size_t)cc * FPAD;

    for (int t = 0; t < LDIM; ++t) {
        float feat = feats[((size_t)b * LDIM + t) * FPAD + cc];
        float mt   = mask[(size_t)b * LDIM + t];

        float best = -3.402823466e38f;
        int   arg  = 0;
        for (int cp = 0; cp < FPAD; cp += 4) {
            v4f sc = *(const v4f*)(s_score + cp);
            v4f tr = *(const v4f*)(trow + cp);
#pragma unroll
            for (int j = 0; j < 4; ++j) {
                float v = sc[j] + tr[j];
                if (v > best) { best = v; arg = cp + j; }
            }
        }
        float ns = (best + feat) * mt + s_score[cc] * (1.0f - mt);

        if (c < CTAGS) s_bp[t * FPAD + c] = (uint8_t)arg;

        __syncthreads();
        if (c < CTAGS) s_score[c] = ns;
        __syncthreads();
    }

    if (c < CTAGS)
        s_score[c] += s_trans[(size_t)STOP_IDX * FPAD + c];
    __syncthreads();

    if (tid == 0) {
        float best = -3.402823466e38f;
        int   tag  = 0;
        for (int i = 0; i < CTAGS; ++i) {
            float v = s_score[i];
            if (v > best) { best = v; tag = i; }
        }
        out[b] = best;

        float* paths = out + BDIM;               // [64, 512]
        int tg = tag;
        paths[(size_t)b * LDIM + (LDIM - 1)] = (float)tg;
        for (int t = LDIM - 1; t >= 1; --t) {
            tg = s_bp[t * FPAD + tg];            // LDS-latency chain
            paths[(size_t)b * LDIM + (t - 1)] = (float)tg;
        }
    }
}

// =====================================================================
extern "C" void kernel_launch(void* const* d_in, const int* in_sizes, int n_in,
                              void* d_out, int out_size, void* d_ws, size_t ws_size,
                              hipStream_t stream) {
    const float* x     = (const float*)d_in[0];   // [64,512,1024]
    const float* mask  = (const float*)d_in[1];   // [64,512]
    const float* W     = (const float*)d_in[2];   // [131,1024]
    const float* bias  = (const float*)d_in[3];   // [131]
    const float* trans = (const float*)d_in[4];   // [131,131]
    float* out = (float*)d_out;                   // 64 + 64*512 floats

    // workspace: [wb bf16][feats]
    const size_t wb_bytes = (size_t)CTAGS * DDIM * sizeof(uint16_t);   // 268 KB
    const size_t wb_pad   = (wb_bytes + 255) & ~(size_t)255;
    uint16_t* wb    = (uint16_t*)d_ws;
    float*    feats = (float*)((char*)d_ws + wb_pad);                  // ~17.3 MB

    crf_convW<<<(CTAGS * DDIM + 255) / 256, 256, 0, stream>>>(W, wb);

    crf_proj_wmma<<<MROWS / (16 * 8), 256, 0, stream>>>(x, wb, bias, feats);

    const size_t smem_bytes =
        (size_t)(CTAGS * FPAD + FPAD) * sizeof(float)   // trans + score
        + (size_t)LDIM * FPAD;                          // u8 backpointers
    crf_viterbi<<<BDIM, 160, smem_bytes, stream>>>(feats, mask, trans, out);
}
